// PPOController_6906307412092
// MI455X (gfx1250) — matrix operand order repaired and checked by reading, not compiled, not run on hardware
//
#include <hip/hip_runtime.h>
#include <math.h>

#define T_STEPS 16
#define N_ACT   8
#define HID     128
#define EMB     32
#define GATES   512                 // 4*HID
#define N_TILES 32                  // GATES/16
#define K_CHUNKS 5                  // 1 chunk for E=32, 4 chunks for H=128
#define FRAG_HALF_PER_LSTM (N_TILES * K_CHUNKS * 32 * 16)   // 81920
#define FRAG_HALF_TOTAL    (2 * FRAG_HALF_PER_LSTM)         // 163840
#define FRAG_BYTES         (FRAG_HALF_TOTAL * 2)            // 327680
#define BT_FLOATS          (2 * GATES)                      // 1024

typedef _Float16 v16h __attribute__((ext_vector_type(16)));
typedef float    v8f  __attribute__((ext_vector_type(8)));

__device__ __forceinline__ float sigf(float x) { return 1.f / (1.f + expf(-x)); }

// ---------------------------------------------------------------------------
// One-time weight pack: f32 row-major W -> f16 WMMA A-fragments in d_ws.
// Fragment flat index: (((ls*32 + tile)*5 + chunk)*32 + lane)*16 + slot
// A-layout (16-bit A 16x32, contiguous k, same order as B): lane L holds row m = L%16;
//   lanes 0-15 : V0..7 -> K 0..15
//   lanes 16-31: V0..7 -> K 16..31
// slot s -> (p = s/2 VGPR, h = s%2 half)
// ---------------------------------------------------------------------------
__global__ void ppo_pack_weights(
    const float* __restrict__ Wih_a, const float* __restrict__ Whh_a,
    const float* __restrict__ bih_a, const float* __restrict__ bhh_a,
    const float* __restrict__ Wih_c, const float* __restrict__ Whh_c,
    const float* __restrict__ bih_c, const float* __restrict__ bhh_c,
    _Float16* __restrict__ frags, float* __restrict__ btot)
{
    int idx = blockIdx.x * blockDim.x + threadIdx.x;
    if (idx < FRAG_HALF_TOTAL) {
        int s    = idx % 16;
        int lane = (idx / 16) % 32;
        int c    = (idx / 512) % K_CHUNKS;
        int tile = (idx / 2560) % N_TILES;
        int ls   = idx / FRAG_HALF_PER_LSTM;
        int row  = tile * 16 + (lane & 15);
        int hi   = lane >> 4;
        int p = s >> 1, h = s & 1;
        int kl = (2 * p + h) + 16 * hi;
        float v;
        if (c == 0) v = (ls ? Wih_c : Wih_a)[row * EMB + kl];
        else        v = (ls ? Whh_c : Whh_a)[row * HID + (c - 1) * 32 + kl];
        frags[idx] = (_Float16)v;
    } else {
        int j = idx - FRAG_HALF_TOTAL;
        if (j < BT_FLOATS) {
            int ls = j / GATES, g = j % GATES;
            btot[j] = ls ? (bih_c[g] + bhh_c[g]) : (bih_a[g] + bhh_a[g]);
        }
    }
}

// ---------------------------------------------------------------------------
// Persistent single-workgroup scan kernel. 512 threads = 16 waves:
//   waves 0-7  -> actor LSTM gate matvec (4 gate tiles each)
//   waves 8-15 -> critic LSTM gate matvec
// Biases are NOT in the WMMA C operand (zero-init chain) — they are added in
// the elementwise cell update from 4 per-thread registers preloaded once.
// ---------------------------------------------------------------------------
__global__ __launch_bounds__(512) void ppo_main(
    const _Float16* __restrict__ frags, const float* __restrict__ btot,
    const int*   __restrict__ old_actions,
    const float* __restrict__ init_input,
    const float* __restrict__ heads_W, const float* __restrict__ heads_b,
    const float* __restrict__ critic_W, const float* __restrict__ critic_b,
    const float* __restrict__ embed,
    float* __restrict__ out)
{
    __shared__ __align__(32) _Float16 sX[EMB];        // current input x_t (f16)
    __shared__ __align__(32) _Float16 sH[2][HID];     // h (f16, WMMA operand)
    __shared__ float sHf[2][HID];                     // h (f32, for heads)
    __shared__ float sG[2][GATES];                    // gate pre-acts (no bias)
    __shared__ float sLogits[N_ACT];

    const int tid  = threadIdx.x;
    const int lane = tid & 31;
    const int wave = tid >> 5;
    const int ls   = wave >> 3;       // 0 = actor, 1 = critic
    const int wsub = wave & 7;        // wave index within LSTM group
    const int hi   = lane >> 4;

    // Persistent weight fragments in VGPRs: 4 tiles x 5 K-chunks = 160 VGPRs.
    v16h wfrag[4][K_CHUNKS];
    #pragma unroll
    for (int tt = 0; tt < 4; ++tt) {
        int tile = wsub * 4 + tt;
        #pragma unroll
        for (int c = 0; c < K_CHUNKS; ++c) {
            const v16h* wp = ((const v16h*)frags) +
                (((ls * N_TILES + tile) * K_CHUNKS + c) * 32 + lane);
            wfrag[tt][c] = *wp;
        }
    }

    // Cell-update threads (actor: tid<128, critic: 256..383) preload their
    // 4 gate biases once — only 4 VGPRs per thread, no spill pressure.
    const bool is_cell = (tid < HID) || (tid >= 256 && tid < 256 + HID);
    float bi = 0.f, bf = 0.f, bg = 0.f, bo = 0.f;
    if (is_cell) {
        int l = (tid >= 256);
        int j = tid & 127;
        const float* bt = btot + l * GATES;
        bi = bt[j]; bf = bt[HID + j]; bg = bt[2 * HID + j]; bo = bt[3 * HID + j];
    }

    // Init x_0 = init_input, h = c = 0.
    if (tid < EMB) sX[tid] = (_Float16)init_input[tid];
    if (tid < 2 * HID) { sH[tid >> 7][tid & 127] = (_Float16)0.f;
                         sHf[tid >> 7][tid & 127] = 0.f; }
    float c_state = 0.f;   // cell state, live only in cell-update threads
    __syncthreads();

    for (int t = 0; t < T_STEPS; ++t) {
        // -------- gates = W_ih x + W_hh h, via zero-init chained WMMA ------
        #pragma unroll
        for (int tt = 0; tt < 4; ++tt) {
            int tile = wsub * 4 + tt;
            v8f acc = (v8f)(0.f);              // inline-0 C, no live-in regs
            #pragma unroll
            for (int c = 0; c < K_CHUNKS; ++c) {
                // B = input vector broadcast across all 16 columns:
                // lane L loads 16 contiguous halves at k0 + 16*(L/16).
                const _Float16* vecp = (c == 0) ? &sX[16 * hi]
                                                : &sH[ls][(c - 1) * 32 + 16 * hi];
                v16h bfrag = *(const v16h*)vecp;
                acc = __builtin_amdgcn_wmma_f32_16x16x32_f16(
                        false, wfrag[tt][c], false, bfrag,
                        (short)0, acc, false, false);
            }
            // D column 0: lane 0 holds m=0..7, lane 16 holds m=8..15.
            if ((lane & 15) == 0) {
                #pragma unroll
                for (int g = 0; g < 8; ++g)
                    sG[ls][tile * 16 + 8 * hi + g] = acc[g];
            }
        }
        __syncthreads();

        // -------- LSTM cell update (+bias here, not in WMMA) ---------------
        if (is_cell) {
            int l = (tid >= 256);
            int j = tid & 127;
            float ig = sG[l][j]           + bi;
            float fg = sG[l][HID + j]     + bf;
            float gg = sG[l][2 * HID + j] + bg;
            float og = sG[l][3 * HID + j] + bo;
            float cs = sigf(fg) * c_state + sigf(ig) * tanhf(gg);
            c_state  = cs;
            float hh = sigf(og) * tanhf(cs);
            sHf[l][j] = hh;
            sH[l][j]  = (_Float16)hh;
        }
        // -------- refresh x_{t+1} = embed[t*A + a_t] (wave 4) --------------
        if (tid >= 128 && tid < 128 + EMB && t + 1 < T_STEPS) {
            int e = tid - 128;
            int a = old_actions[t];
            sX[e] = (_Float16)embed[(t * N_ACT + a) * EMB + e];
        }
        __syncthreads();

        // -------- actor head logits (lanes 0..7 of wave 0) -----------------
        if (tid < N_ACT) {
            float d = heads_b[t * N_ACT + tid];
            const float* hw = heads_W + (t * N_ACT + tid) * HID;
            for (int k = 0; k < HID; ++k) d += hw[k] * sHf[0][k];
            sLogits[tid] = d;
        }
        // -------- critic value (thread 256) --------------------------------
        if (tid == 256) {
            float v = critic_b[0];
            for (int k = 0; k < HID; ++k) v += critic_W[k] * sHf[1][k];
            out[2 * T_STEPS + t] = v;
        }
        __syncthreads();

        // -------- log-softmax, log-prob, entropy (thread 0) ----------------
        if (tid == 0) {
            float m = sLogits[0];
            for (int a = 1; a < N_ACT; ++a) m = fmaxf(m, sLogits[a]);
            float se = 0.f;
            for (int a = 0; a < N_ACT; ++a) se += expf(sLogits[a] - m);
            float lse = m + logf(se);
            float dot = 0.f;
            for (int a = 0; a < N_ACT; ++a) dot += expf(sLogits[a] - lse) * sLogits[a];
            out[t]           = sLogits[old_actions[t]] - lse;  // log-prob
            out[T_STEPS + t] = lse - dot;                      // entropy
        }
        // Next iteration's WMMA reads sX/sH written before barrier 2 above;
        // sG/sLogits reuse is fenced by the barriers inside the next step.
    }
}

extern "C" void kernel_launch(void* const* d_in, const int* in_sizes, int n_in,
                              void* d_out, int out_size, void* d_ws, size_t ws_size,
                              hipStream_t stream)
{
    const int*   old_actions = (const int*)  d_in[0];
    const float* init_input  = (const float*)d_in[1];
    const float* Wih_a   = (const float*)d_in[2];
    const float* Whh_a   = (const float*)d_in[3];
    const float* bih_a   = (const float*)d_in[4];
    const float* bhh_a   = (const float*)d_in[5];
    const float* headsW  = (const float*)d_in[6];
    const float* headsb  = (const float*)d_in[7];
    const float* Wih_c   = (const float*)d_in[8];
    const float* Whh_c   = (const float*)d_in[9];
    const float* bih_c   = (const float*)d_in[10];
    const float* bhh_c   = (const float*)d_in[11];
    const float* criticW = (const float*)d_in[12];
    const float* criticb = (const float*)d_in[13];
    const float* embed   = (const float*)d_in[14];

    _Float16* frags = (_Float16*)d_ws;
    float*    btot  = (float*)((char*)d_ws + FRAG_BYTES);

    int n1 = FRAG_HALF_TOTAL + BT_FLOATS;
    ppo_pack_weights<<<(n1 + 255) / 256, 256, 0, stream>>>(
        Wih_a, Whh_a, bih_a, bhh_a, Wih_c, Whh_c, bih_c, bhh_c, frags, btot);

    ppo_main<<<1, 512, 0, stream>>>(
        frags, btot, old_actions, init_input,
        headsW, headsb, criticW, criticb, embed, (float*)d_out);
}
